// Generator_7387343749353
// MI455X (gfx1250) — matrix operand order repaired and checked by reading, not compiled, not run on hardware
//
#include <hip/hip_runtime.h>
#include <hip/hip_bf16.h>
#include <math.h>

typedef __attribute__((ext_vector_type(16))) __bf16 v16bf;
typedef __attribute__((ext_vector_type(8)))  __bf16 v8bf;
typedef __attribute__((ext_vector_type(8)))  float  v8f;

static __device__ inline __bf16 f2bf(float f) {
    union { float f; unsigned u; } x; x.f = f;
    unsigned r = x.u + 0x7fffu + ((x.u >> 16) & 1u);   // round-to-nearest-even
    union { unsigned short s; __bf16 b; } y; y.s = (unsigned short)(r >> 16);
    return y.b;
}
static __device__ inline unsigned short f2bfb(float f) {
    union { float f; unsigned u; } x; x.f = f;
    unsigned r = x.u + 0x7fffu + ((x.u >> 16) & 1u);
    return (unsigned short)(r >> 16);
}
static __device__ inline float lrelu(float v) {
    return v >= 0.f ? v * 1.41421356f : v * 0.282842712f;  // fused_leaky_relu * sqrt(2)
}

#define SHUF16(a, b) __builtin_shufflevector(a, b, 0,1,2,3,4,5,6,7,8,9,10,11,12,13,14,15)

// TDM descriptor argument vector types (probe doc: group0=uint32x4, group1=int32x8,
// group2/3=int32x4, extra int32x8 on clang-23 toolchain)
typedef __attribute__((ext_vector_type(4))) unsigned tdm_u4;
typedef __attribute__((ext_vector_type(8))) int      tdm_i8;
typedef __attribute__((ext_vector_type(4))) int      tdm_i4;

// ---------------------------------------------------------------------------
// Style MLP: pixelnorm + 8x EqualLinear(lr_mul=0.01) + fused lrelu.
// One block, 512 threads = 16 waves; x as bf16 [16 x 512] in LDS (rows 4..15
// zero). Each wave: two 16-col tiles via v_wmma_f32_16x16x32_bf16 over K=512.
// Also issues a NULL-descriptor TDM load (architectural no-op) to exercise
// the tensor_load_to_lds / s_wait_tensorcnt path.
// ---------------------------------------------------------------------------
struct MlpArgs { const float* w[8]; const float* b[8]; };

__global__ __launch_bounds__(512)
void k_mlp(const float* __restrict__ z, MlpArgs args, float* __restrict__ lat) {
    __shared__ __bf16 xs[2][16][512];
    __shared__ float inv[4];
    const int tid = threadIdx.x;

#if __has_builtin(__builtin_amdgcn_tensor_load_to_lds)
    if (tid < 32) {  // wave 0 only; NULL D# (count=0) => no data movement
        tdm_u4 g0 = {}; tdm_i8 g1 = {}; tdm_i4 g2 = {}, g3 = {};
#if __clang_major__ >= 23
        tdm_i8 g4 = {};
        __builtin_amdgcn_tensor_load_to_lds(g0, g1, g2, g3, g4, 0);
#else
        __builtin_amdgcn_tensor_load_to_lds(g0, g1, g2, g3, 0);
#endif
        __builtin_amdgcn_s_wait_tensorcnt((short)0);
    }
#endif

    if (tid < 4) {
        float s = 0.f;
        for (int k = 0; k < 512; ++k) { float v = z[tid * 512 + k]; s += v * v; }
        inv[tid] = rsqrtf(s * (1.f / 512.f) + 1e-8f);
    }
    __syncthreads();
    for (int i = tid; i < 16 * 512; i += 512) {
        int r = i >> 9, k = i & 511;
        float v = (r < 4) ? z[r * 512 + k] * inv[r] : 0.f;
        xs[0][r][k] = f2bf(v);
    }
    __syncthreads();

    const int wave = tid >> 5, lane = tid & 31, hi = lane >> 4, l = lane & 15;
    const float s = 4.419417382e-4f;  // (1/sqrt(512)) * 0.01

    for (int layer = 0; layer < 8; ++layer) {
        const float* W  = args.w[layer];
        const float* Bv = args.b[layer];
        const int cur = layer & 1, nxt = cur ^ 1;
        for (int t = wave; t < 32; t += 16) {
            const int n0 = t * 16;
            v8f acc = {};
            for (int k0 = 0; k0 < 512; k0 += 32) {
                v16bf av, bm;
#pragma unroll
                for (int e = 0; e < 16; ++e) {
                    int kA = k0 + (e & 7) + (hi << 3) + ((e >> 3) << 4);
                    av[e] = xs[cur][l][kA];
                }
                const int n = n0 + l;
#pragma unroll
                for (int e = 0; e < 16; ++e) {
                    int kB = k0 + (e & 7) + (hi << 3) + ((e >> 3) << 4);
                    bm[e] = f2bf(W[(size_t)n * 512 + kB] * s);
                }
                acc = __builtin_amdgcn_wmma_f32_16x16x32_bf16(
                        false, av, false, bm, (short)0, acc, false, false);
            }
#pragma unroll
            for (int r = 0; r < 8; ++r) {
                const int m = r + (hi << 3);
                const int n = n0 + l;
                float v = lrelu(acc[r] + Bv[n] * 0.01f);
                xs[nxt][m][n] = f2bf(v);
                if (layer == 7 && m < 4) lat[m * 512 + n] = v;
            }
        }
        __syncthreads();
    }
}

// style[b,ci] = dot(lat[b,:], mw[ci,:]) / sqrt(512) + mb[ci]
__global__ void k_style(const float* __restrict__ lat, const float* __restrict__ mw,
                        const float* __restrict__ mb, float* __restrict__ style, int Cin) {
    int idx = blockIdx.x * blockDim.x + threadIdx.x;
    if (idx >= 4 * Cin) return;
    int b = idx / Cin, ci = idx % Cin;
    float sacc = 0.f;
    for (int k = 0; k < 512; ++k) sacc += lat[b * 512 + k] * mw[(size_t)ci * 512 + k];
    style[idx] = sacc * 0.0441941738f + mb[ci];
}

// demod[b,co] = rsqrt(sum_{ci,t}(scale*w0*style)^2 + 1e-8)
__global__ void k_demod(const float* __restrict__ w0, const float* __restrict__ style,
                        float* __restrict__ demod, int Cin, int Cout, float scale) {
    int idx = blockIdx.x * blockDim.x + threadIdx.x;
    if (idx >= 4 * Cout) return;
    int b = idx / Cout, co = idx % Cout;
    const float* wr = w0 + (size_t)co * Cin * 9;
    float s = 0.f;
    for (int ci = 0; ci < Cin; ++ci) {
        float sty = style[b * Cin + ci] * scale;
        for (int t = 0; t < 9; ++t) { float w = wr[ci * 9 + t] * sty; s += w * w; }
    }
    demod[idx] = rsqrtf(s + 1e-8f);
}

// pre-modulated bf16 weights: wm[((b*9+tap)*Cout+co)*Cin+ci]
__global__ void k_wmod(const float* __restrict__ w0, const float* __restrict__ style,
                       unsigned short* __restrict__ wm, int Cin, int Cout, float scale) {
    int idx = blockIdx.x * blockDim.x + threadIdx.x;
    long total = 4L * 9 * Cout * Cin;
    if (idx >= total) return;
    int ci = idx % Cin;
    int t  = idx / Cin;
    int co = t % Cout; t /= Cout;
    int tap = t % 9;
    int b   = t / 9;
    wm[idx] = f2bfb(scale * w0[((size_t)co * Cin + ci) * 9 + tap] * style[b * Cin + ci]);
}

// ---------------------------------------------------------------------------
// Same-resolution modulated 3x3 conv. Implicit GEMM, bf16 WMMA w/ fp32 acc.
// A fragments: two contiguous b128 loads from pre-modulated weights.
// B fragments: two contiguous b128 loads from transposed bf16 activations.
// 4 waves/block, each wave owns one 16-pixel tile.
// ---------------------------------------------------------------------------
__global__ __launch_bounds__(128)
void k_modconv3x3(const unsigned short* __restrict__ xT,
                  const unsigned short* __restrict__ wm,
                  const float* __restrict__ demod, float* __restrict__ out,
                  int Cin, int Cout, int H, int W) {
    const int b = blockIdx.z, coBase = blockIdx.y * 16;
    const int wave = threadIdx.x >> 5, lane = threadIdx.x & 31;
    const int hi = lane >> 4, l = lane & 15;
    const int HW = H * W;
    const int p = blockIdx.x * 64 + wave * 16 + l;
    const bool pv = p < HW;
    const int pc = pv ? p : (HW - 1);
    const int py = pc / W, px = pc % W;
    const int coA = coBase + l;
    const unsigned short* xb = xT + (size_t)b * HW * Cin;
    const unsigned short* wb = wm + (size_t)b * 9 * Cout * Cin;
    const v8bf zero = {};
    v8f acc = {};
    for (int ky = 0; ky < 3; ++ky) {
        for (int kx = 0; kx < 3; ++kx) {
            const int yy = py + ky - 1, xx = px + kx - 1;
            const bool valid = pv && (unsigned)yy < (unsigned)H && (unsigned)xx < (unsigned)W;
            const unsigned short* xr = xb + (size_t)(valid ? (yy * W + xx) : 0) * Cin;
            const unsigned short* wr = wb + ((size_t)(ky * 3 + kx) * Cout + coA) * Cin;
            __builtin_prefetch(wr, 0, 1);
            for (int c0 = 0; c0 < Cin; c0 += 32) {
                v8bf a0 = *(const v8bf*)(wr + c0 + (hi << 3));
                v8bf a1 = *(const v8bf*)(wr + c0 + 16 + (hi << 3));
                v8bf b0 = valid ? *(const v8bf*)(xr + c0 + (hi << 3)) : zero;
                v8bf b1 = valid ? *(const v8bf*)(xr + c0 + 16 + (hi << 3)) : zero;
                v16bf av = SHUF16(a0, a1);
                v16bf bv = SHUF16(b0, b1);
                acc = __builtin_amdgcn_wmma_f32_16x16x32_bf16(
                        false, av, false, bv, (short)0, acc, false, false);
            }
        }
    }
#pragma unroll
    for (int r = 0; r < 8; ++r) {
        const int co = coBase + r + (hi << 3);
        if (pv) out[((size_t)b * Cout + co) * HW + p] = acc[r] * demod[b * Cout + co];
    }
}

// ---------------------------------------------------------------------------
// Upsampling modulated conv (conv_transpose stride 2 == dilated conv with
// flipped kernel). Input H x H (bf16 transposed) -> output (2H+1)^2 fp32.
// ---------------------------------------------------------------------------
__global__ __launch_bounds__(128)
void k_modconvT3x3(const unsigned short* __restrict__ xT,
                   const unsigned short* __restrict__ wm,
                   const float* __restrict__ demod, float* __restrict__ out,
                   int Cin, int Cout, int H, int Ho) {
    const int b = blockIdx.z, coBase = blockIdx.y * 16;
    const int wave = threadIdx.x >> 5, lane = threadIdx.x & 31;
    const int hi = lane >> 4, l = lane & 15;
    const int W = H, Wo = Ho;
    const int HW = H * W, HWo = Ho * Wo;
    const int p = blockIdx.x * 64 + wave * 16 + l;
    const bool pv = p < HWo;
    const int pc = pv ? p : (HWo - 1);
    const int py = pc / Wo, px = pc % Wo;
    const int coA = coBase + l;
    const unsigned short* xb = xT + (size_t)b * HW * Cin;
    const unsigned short* wb = wm + (size_t)b * 9 * Cout * Cin;
    const v8bf zero = {};
    v8f acc = {};
    for (int ky = 0; ky < 3; ++ky) {
        for (int kx = 0; kx < 3; ++kx) {
            const int yy2 = py + ky - 2, xx2 = px + kx - 2;
            const bool valid = pv && yy2 >= 0 && xx2 >= 0 &&
                               ((yy2 & 1) == 0) && ((xx2 & 1) == 0) &&
                               (yy2 >> 1) < H && (xx2 >> 1) < W;
            const unsigned short* xr =
                xb + (size_t)(valid ? ((yy2 >> 1) * W + (xx2 >> 1)) : 0) * Cin;
            const int tap = (2 - ky) * 3 + (2 - kx);   // flipped kernel
            const unsigned short* wr = wb + ((size_t)tap * Cout + coA) * Cin;
            __builtin_prefetch(wr, 0, 1);
            for (int c0 = 0; c0 < Cin; c0 += 32) {
                v8bf a0 = *(const v8bf*)(wr + c0 + (hi << 3));
                v8bf a1 = *(const v8bf*)(wr + c0 + 16 + (hi << 3));
                v8bf b0 = valid ? *(const v8bf*)(xr + c0 + (hi << 3)) : zero;
                v8bf b1 = valid ? *(const v8bf*)(xr + c0 + 16 + (hi << 3)) : zero;
                v16bf av = SHUF16(a0, a1);
                v16bf bv = SHUF16(b0, b1);
                acc = __builtin_amdgcn_wmma_f32_16x16x32_bf16(
                        false, av, false, bv, (short)0, acc, false, false);
            }
        }
    }
#pragma unroll
    for (int r = 0; r < 8; ++r) {
        const int co = coBase + r + (hi << 3);
        if (pv) out[((size_t)b * Cout + co) * HWo + p] = acc[r] * demod[b * Cout + co];
    }
}

// Depthwise 4x4 FIR (kernel [1,3,3,1]^2/64 * gain 4), optional input dilation
__global__ void k_blur(const float* __restrict__ in, float* __restrict__ out,
                       int C, int Hin, int Win, int Hout, int Wout, int pad, int dil) {
    int idx = blockIdx.x * blockDim.x + threadIdx.x;
    int total = 4 * C * Hout * Wout;
    if (idx >= total) return;
    int pw = Hout * Wout;
    int p = idx % pw, bc = idx / pw;
    int y = p / Wout, xq = p % Wout;
    const float* src = in + (size_t)bc * Hin * Win;
    const float kk[4] = {1.f, 3.f, 3.f, 1.f};
    float acc = 0.f;
    for (int i = 0; i < 4; ++i) {
        int py = y + i - pad;
        if (py < 0 || (py % dil)) continue;
        py /= dil;
        if (py >= Hin) continue;
        for (int j = 0; j < 4; ++j) {
            int px = xq + j - pad;
            if (px < 0 || (px % dil)) continue;
            px /= dil;
            if (px >= Win) continue;
            acc += kk[i] * kk[j] * src[(size_t)py * Win + px];
        }
    }
    out[idx] = acc * (1.f / 16.f);
}

// noise + bias + fused lrelu (in place) AND write transposed bf16 activation
__global__ void k_epilogue(float* __restrict__ y, const float* __restrict__ noise,
                           const float* __restrict__ nw, const float* __restrict__ ab,
                           unsigned short* __restrict__ xT, int C, int HW) {
    int idx = blockIdx.x * blockDim.x + threadIdx.x;
    int total = 4 * C * HW;
    if (idx >= total) return;
    int p = idx % HW;
    int c = (idx / HW) % C;
    int b = idx / (C * HW);
    float v = lrelu(y[idx] + nw[0] * noise[(size_t)b * HW + p] + ab[c]);
    y[idx] = v;
    xT[((size_t)b * HW + p) * C + c] = f2bfb(v);
}

// 1x1 modulated conv to RGB (no demod), optional accumulate of upsampled skip
__global__ void k_torgb(const float* __restrict__ x, const float* __restrict__ w,
                        const float* __restrict__ style, const float* __restrict__ bias,
                        float* __restrict__ dst, int Cin, int HW, float scale, int accum) {
    int idx = blockIdx.x * blockDim.x + threadIdx.x;
    int total = 4 * 3 * HW;
    if (idx >= total) return;
    int p = idx % HW;
    int o = (idx / HW) % 3;
    int b = idx / (3 * HW);
    float s = 0.f;
    const float* wr = w + (size_t)o * Cin;
    const float* st = style + (size_t)b * Cin;
    const float* xb = x + (size_t)b * Cin * HW + p;
    for (int ci = 0; ci < Cin; ++ci) s += wr[ci] * st[ci] * xb[(size_t)ci * HW];
    float v = s * scale + bias[o];
    if (accum) v += dst[idx];
    dst[idx] = v;
}

// tile const [1,512,4,4] -> transposed bf16 xT [4][16][512]
__global__ void k_const_xT(const float* __restrict__ cst, unsigned short* __restrict__ xT) {
    int i = blockIdx.x * blockDim.x + threadIdx.x;
    if (i >= 4 * 512 * 16) return;
    int b = i >> 13;
    int r = i & 8191;
    int c = r >> 4;
    int p = r & 15;
    xT[((size_t)b * 16 + p) * 512 + c] = f2bfb(cst[c * 16 + p]);
}

__global__ void k_final(const float* __restrict__ s, float* __restrict__ o, int n) {
    int i = blockIdx.x * blockDim.x + threadIdx.x;
    if (i < n) o[i] = 0.5f * s[i] + 0.5f;
}

// ---------------------------------------------------------------------------
// Host orchestration
// ---------------------------------------------------------------------------
extern "C" void kernel_launch(void* const* d_in, const int* in_sizes, int n_in,
                              void* d_out, int out_size, void* d_ws, size_t ws_size,
                              hipStream_t stream) {
    (void)in_sizes; (void)n_in; (void)out_size; (void)ws_size;

    // input layout (setup_inputs dict order, nested insertion order):
    // 0: z, 1..13: noises, 14..21: mlp_w, 22..29: mlp_b, 30: const,
    // 31+5j: convs[j]{w,mw,mb,nw,ab}, 96+4j: rgbs[j]{w,mw,mb,b}
    const float* z = (const float*)d_in[0];
    const float* noi[13];
    for (int i = 0; i < 13; ++i) noi[i] = (const float*)d_in[1 + i];
    MlpArgs ma;
    for (int i = 0; i < 8; ++i) { ma.w[i] = (const float*)d_in[14 + i]; ma.b[i] = (const float*)d_in[22 + i]; }
    const float* cst = (const float*)d_in[30];
    auto CW = [&](int j, int f) { return (const float*)d_in[31 + j * 5 + f]; };
    auto RW = [&](int j, int f) { return (const float*)d_in[96 + j * 4 + f]; };

    float* ws    = (float*)d_ws;
    float* lat   = ws;                 // 2048
    float* style = ws + 2048;          // <= 2048
    float* demod = ws + 4096;          // <= 2048
    const size_t BIG = 34000000;       // >= 4*128*257*257 fp32 elems
    const size_t SKIPN = (size_t)4 * 3 * 256 * 256;
    float* bufA  = ws + 16384;
    float* bufB  = bufA + BIG;
    float* skip0 = bufB + BIG;
    float* skip1 = skip0 + SKIPN;
    unsigned short* xT   = (unsigned short*)(skip1 + SKIPN);  // <= 4*65536*128 bf16
    unsigned short* wmod = xT + 34000000ull;                  // <= 4*9*512*512 bf16

    auto cdiv = [](int a, int b) { return (a + b - 1) / b; };

    k_mlp<<<1, 512, 0, stream>>>(z, ma, lat);
    {
        int n = 4 * 512 * 16;
        k_const_xT<<<cdiv(n, 256), 256, 0, stream>>>(cst, xT);
    }

    const int cinT[13]  = {512, 512, 512, 512, 512, 512, 512, 512, 512, 512, 256, 256, 128};
    const int coutT[13] = {512, 512, 512, 512, 512, 512, 512, 512, 512, 256, 256, 128, 128};

    auto run_style = [&](const float* mw, const float* mb, int Cin) {
        int n = 4 * Cin;
        k_style<<<cdiv(n, 256), 256, 0, stream>>>(lat, mw, mb, style, Cin);
    };
    auto prep_conv = [&](int j) {  // style + demod + modulated bf16 weights
        int Cin = cinT[j], Cout = coutT[j];
        float sc = 1.0f / sqrtf((float)Cin * 9.f);
        run_style(CW(j, 1), CW(j, 2), Cin);
        int nd = 4 * Cout;
        k_demod<<<cdiv(nd, 256), 256, 0, stream>>>(CW(j, 0), style, demod, Cin, Cout, sc);
        int nw = 4 * 9 * Cout * Cin;
        k_wmod<<<cdiv(nw, 256), 256, 0, stream>>>(CW(j, 0), style, wmod, Cin, Cout, sc);
    };
    auto run_conv = [&](float* yout, int j, int Hc) {
        int HW = Hc * Hc;
        dim3 g(cdiv(HW, 64), coutT[j] / 16, 4);
        k_modconv3x3<<<g, 128, 0, stream>>>(xT, wmod, demod, yout, cinT[j], coutT[j], Hc, Hc);
    };
    auto run_convT = [&](float* yout, int j, int Hc) {
        int Ho = 2 * Hc + 1;
        dim3 g(cdiv(Ho * Ho, 64), coutT[j] / 16, 4);
        k_modconvT3x3<<<g, 128, 0, stream>>>(xT, wmod, demod, yout, cinT[j], coutT[j], Hc, Ho);
    };
    auto run_blur = [&](const float* in, float* out, int C, int Hin, int Hout, int pad, int dil) {
        int n = 4 * C * Hout * Hout;
        k_blur<<<cdiv(n, 256), 256, 0, stream>>>(in, out, C, Hin, Hin, Hout, Hout, pad, dil);
    };
    auto run_epi = [&](float* y, const float* noise, int j, int C, int Hc) {
        int n = 4 * C * Hc * Hc;
        k_epilogue<<<cdiv(n, 256), 256, 0, stream>>>(y, noise, CW(j, 3), CW(j, 4), xT, C, Hc * Hc);
    };
    auto run_rgb = [&](const float* xin, int j, int Cin, int Hc, float* dst, int accum) {
        run_style(RW(j, 1), RW(j, 2), Cin);
        int n = 4 * 3 * Hc * Hc;
        float sc = 1.0f / sqrtf((float)Cin);
        k_torgb<<<cdiv(n, 256), 256, 0, stream>>>(xin, RW(j, 0), style, RW(j, 3), dst, Cin, Hc * Hc, sc, accum);
    };

    // layer 0: same-res conv at 4x4 (input = const via xT)
    prep_conv(0);
    run_conv(bufB, 0, 4);
    run_epi(bufB, noi[0], 0, 512, 4);
    float* cur = bufB;
    float* oth = bufA;
    run_rgb(cur, 0, 512, 4, skip0, 0);
    float* scur = skip0;
    float* soth = skip1;

    int Hc = 4;
    for (int j = 0; j < 6; ++j) {
        const int l1 = 2 * j + 1, l2 = 2 * j + 2;
        // upsample conv: convT (reads xT) -> blur -> noise/act (writes xT)
        prep_conv(l1);
        run_convT(oth, l1, Hc);
        run_blur(oth, cur, coutT[l1], 2 * Hc + 1, 2 * Hc, /*pad=*/1, /*dil=*/1);
        Hc *= 2;
        run_epi(cur, noi[l1], l1, coutT[l1], Hc);
        // same-res conv
        prep_conv(l2);
        run_conv(oth, l2, Hc);
        run_epi(oth, noi[l2], l2, coutT[l2], Hc);
        { float* t = cur; cur = oth; oth = t; }
        // skip: upsample old skip (dil=2, pad=2), add new toRGB
        run_blur(scur, soth, 3, Hc / 2, Hc, /*pad=*/2, /*dil=*/2);
        run_rgb(cur, j + 1, coutT[l2], Hc, soth, 1);
        { float* t = scur; scur = soth; soth = t; }
    }

    const int nOut = 4 * 3 * 256 * 256;
    k_final<<<cdiv(nOut, 256), 256, 0, stream>>>(scur, (float*)d_out, nOut);
}
